// DisentangledAttention_35716948033732
// MI455X (gfx1250) — matrix operand order repaired and checked by reading, not compiled, not run on hardware
//
#include <hip/hip_runtime.h>
#include <cstdint>
#include <cstddef>

// ---------------------------------------------------------------------------
// DisentangledAttention for MI455X (gfx1250, wave32, WMMA bf16 16x16x32)
//   B=8, S=4096, HID=1024, NH=16, DH=64  ->  M=32768, N=K=1024
//   Pipeline:
//     1) 5x GEMM (fp32 in -> bf16 out in ws):  cq, ck, cv, pq, pk
//     2) per-token 16x16 head-attention kernel (8 WMMAs/token) -> bf16 ws
//     3) 1x GEMM (bf16 in -> fp32 out): final output projection
//   GEMM: block tile 128x256, 8 waves (2x4), wave tile 64x64 => 16 WMMA per
//   K-step of 32 against 16 LDS b128 fragment loads.
// ---------------------------------------------------------------------------

typedef __attribute__((ext_vector_type(16))) __bf16 v16bf;
typedef __attribute__((ext_vector_type(8)))  float  v8f;

#define BM  128
#define BN  256
#define BK  32
#define LDK 40   // padded bf16 K-stride: 80 bytes, keeps 16B alignment

struct Q2 { uint4 lo, hi; };

static __device__ inline v16bf make_frag(uint4 lo, uint4 hi) {
  Q2 q{lo, hi};
  return __builtin_bit_cast(v16bf, q);
}

// fp32 -> bf16 via native converts (v_cvt_*_bf16_f32 class ops on gfx1250)
static __device__ inline unsigned short bf_bits(float f) {
  return __builtin_bit_cast(unsigned short, (__bf16)f);
}
static __device__ inline unsigned pk2(float a, float b) {
  return (unsigned)bf_bits(a) | ((unsigned)bf_bits(b) << 16);
}

// ---------------------------------------------------------------------------
// GEMM:  Y[M,N] = X[M,K] @ W[N,K]^T + bias[N]
// ---------------------------------------------------------------------------
template<bool IN_BF16, bool OUT_F32>
__global__ __launch_bounds__(256) void gemm_bias_kernel(
    const void* __restrict__ Xv, const float* __restrict__ W,
    const float* __restrict__ bias, void* __restrict__ Yv,
    int M, int N, int K)
{
  __shared__ __align__(16) unsigned short As[BM * LDK];   // 10.0 KB
  __shared__ __align__(16) unsigned short Bs[BN * LDK];   // 20.0 KB

  const int tid  = threadIdx.x;
  const int lane = tid & 31;
  const int wave = tid >> 5;
  const int wm   = wave >> 2;   // 0..1  (M direction, 64 rows each)
  const int wn   = wave & 3;    // 0..3  (N direction, 64 cols each)
  const int half = lane >> 4;   // lane half selects K sub-block per ISA layout
  const int l16  = lane & 15;

  const int block_n = blockIdx.x * BN;
  const int block_m = blockIdx.y * BM;

  // cooperative staging: thread -> (row = tid>>1, 16 cols at (tid&1)*16)
  const int srow = tid >> 1;
  const int scol = (tid & 1) * 16;

  v8f acc[4][4] = {};

  for (int k0 = 0; k0 < K; k0 += BK) {
    // ---- stage A (activations): 128 rows x 32 cols, bf16 in LDS
    {
      uint4* dst = (uint4*)&As[srow * LDK + scol];
      if constexpr (IN_BF16) {
        const uint4* src = (const uint4*)((const unsigned short*)Xv +
                           (size_t)(block_m + srow) * K + (k0 + scol));
        dst[0] = src[0]; dst[1] = src[1];
      } else {
        const float4* src = (const float4*)((const float*)Xv +
                            (size_t)(block_m + srow) * K + (k0 + scol));
        float4 a = src[0], b = src[1], c = src[2], d = src[3];
        dst[0] = make_uint4(pk2(a.x,a.y), pk2(a.z,a.w), pk2(b.x,b.y), pk2(b.z,b.w));
        dst[1] = make_uint4(pk2(c.x,c.y), pk2(c.z,c.w), pk2(d.x,d.y), pk2(d.z,d.w));
      }
    }
    // ---- stage B = rows of W (W is [N,K] row-major): 256 rows x 32 cols
#pragma unroll
    for (int rr = 0; rr < 2; ++rr) {
      const int r = srow + rr * 128;
      uint4* dst = (uint4*)&Bs[r * LDK + scol];
      const float4* src = (const float4*)(W + (size_t)(block_n + r) * K + (k0 + scol));
      float4 a = src[0], b = src[1], c = src[2], d = src[3];
      dst[0] = make_uint4(pk2(a.x,a.y), pk2(a.z,a.w), pk2(b.x,b.y), pk2(b.z,b.w));
      dst[1] = make_uint4(pk2(c.x,c.y), pk2(c.z,c.w), pk2(d.x,d.y), pk2(d.z,d.w));
    }
    // prefetch next K tile into cache (global_prefetch_b8)
    if (k0 + BK < K) {
      if constexpr (IN_BF16)
        __builtin_prefetch((const unsigned short*)Xv + (size_t)(block_m + srow) * K + (k0 + BK + scol));
      else
        __builtin_prefetch((const float*)Xv + (size_t)(block_m + srow) * K + (k0 + BK + scol));
      __builtin_prefetch(W + (size_t)(block_n + srow) * K + (k0 + BK + scol));
      __builtin_prefetch(W + (size_t)(block_n + srow + 128) * K + (k0 + BK + scol));
    }
    __syncthreads();

    // ---- fragment loads per ISA 7.12.2 (16-bit A 16x32 / B 32x16 layouts)
    v16bf af[4], bfr[4];
#pragma unroll
    for (int mi = 0; mi < 4; ++mi) {
      // A: lane holds row M=l16; elems 0..7 -> K=half*8+e, elems 8..15 -> K=16+half*8+e
      const unsigned short* p = &As[(wm * 64 + mi * 16 + l16) * LDK + half * 8];
      af[mi] = make_frag(*(const uint4*)p, *(const uint4*)(p + 16));
    }
#pragma unroll
    for (int ni = 0; ni < 4; ++ni) {
      // B: lane holds col N=l16; elems 0..7 -> K=half*8+e, elems 8..15 -> K=16+half*8+e
      const unsigned short* p = &Bs[(wn * 64 + ni * 16 + l16) * LDK + half * 8];
      bfr[ni] = make_frag(*(const uint4*)p, *(const uint4*)(p + 16));
    }
#pragma unroll
    for (int mi = 0; mi < 4; ++mi)
#pragma unroll
      for (int ni = 0; ni < 4; ++ni)
        acc[mi][ni] = __builtin_amdgcn_wmma_f32_16x16x32_bf16(
            false, af[mi], false, bfr[ni], (short)0, acc[mi][ni], false, false);
    __syncthreads();
  }

  // ---- epilogue: f32 C/D layout: lane = N=l16, VGPR v -> M = v + 8*half
#pragma unroll
  for (int ni = 0; ni < 4; ++ni) {
    const int n = block_n + wn * 64 + ni * 16 + l16;
    const float bv = bias[n];
#pragma unroll
    for (int mi = 0; mi < 4; ++mi) {
      const int m0 = block_m + wm * 64 + mi * 16 + half * 8;
#pragma unroll
      for (int v = 0; v < 8; ++v) {
        const float r = acc[mi][ni][v] + bv;
        const size_t off = (size_t)(m0 + v) * N + n;
        if constexpr (OUT_F32) ((float*)Yv)[off] = r;
        else                   ((unsigned short*)Yv)[off] = bf_bits(r);
      }
    }
  }
}

// ---------------------------------------------------------------------------
// Per-token head attention: one wave per token.
//   S[g,h] = sum_d ck[g,d]*cq[h,d] + pk[g,d]*pq[h,d]   (4 WMMAs, M=g, N=h)
//   P = softmax over g (M axis) -> accumulator layout == A-fragment layout,
//   so P feeds directly into  out[h,d] = P^T(16x16,K-padded) x V(16x64).
// ---------------------------------------------------------------------------
__global__ __launch_bounds__(256) void attn_kernel(
    const unsigned short* __restrict__ cq, const unsigned short* __restrict__ ck,
    const unsigned short* __restrict__ cv, const unsigned short* __restrict__ pq,
    const unsigned short* __restrict__ pk, unsigned short* __restrict__ out)
{
  __shared__ __align__(16) unsigned short vt[8 * 64 * 16];  // per-wave V^T [d][g]
  const int tid  = threadIdx.x;
  const int lane = tid & 31;
  const int wave = tid >> 5;
  const int l16  = lane & 15;
  const int half = lane >> 4;
  const size_t rowoff = ((size_t)blockIdx.x * 8 + wave) * 1024;

  // stage V transposed: vt[d*16+g] = cv[g*64+d]
  unsigned short* myvt = &vt[wave * 1024];
  const unsigned short* cvr = cv + rowoff;
#pragma unroll
  for (int t = 0; t < 32; ++t) {
    const int i = t * 32 + lane;
    myvt[(i & 63) * 16 + (i >> 6)] = cvr[i];
  }
  __syncthreads();

  const unsigned short* ckr = ck + rowoff;
  const unsigned short* cqr = cq + rowoff;
  const unsigned short* pkr = pk + rowoff;
  const unsigned short* pqr = pq + rowoff;

  auto loadA = [&](const unsigned short* r, int dbase) {   // row = l16, K per-layout
    const unsigned short* p = r + l16 * 64 + dbase + half * 8;
    return make_frag(*(const uint4*)p, *(const uint4*)(p + 16));
  };
  auto loadB = [&](const unsigned short* r, int dbase) {   // col = l16, K per-layout
    const unsigned short* p = r + l16 * 64 + dbase + half * 8;
    return make_frag(*(const uint4*)p, *(const uint4*)(p + 16));
  };

  v8f s = {};
  s = __builtin_amdgcn_wmma_f32_16x16x32_bf16(false, loadA(ckr, 0),  false, loadB(cqr, 0),  (short)0, s, false, false);
  s = __builtin_amdgcn_wmma_f32_16x16x32_bf16(false, loadA(ckr, 32), false, loadB(cqr, 32), (short)0, s, false, false);
  s = __builtin_amdgcn_wmma_f32_16x16x32_bf16(false, loadA(pkr, 0),  false, loadB(pqr, 0),  (short)0, s, false, false);
  s = __builtin_amdgcn_wmma_f32_16x16x32_bf16(false, loadA(pkr, 32), false, loadB(pqr, 32), (short)0, s, false, false);

  // softmax over g (= M axis): lane holds g = v + 8*half for fixed h = l16;
  // the other 8 g-values live in lane^16 -> one shuffle.
  float mx = -3.0e38f;
#pragma unroll
  for (int v = 0; v < 8; ++v) { s[v] *= 0.125f; mx = fmaxf(mx, s[v]); }
  mx = fmaxf(mx, __shfl_xor(mx, 16, 32));
  float p8[8]; float sum = 0.f;
#pragma unroll
  for (int v = 0; v < 8; ++v) { p8[v] = __expf(s[v] - mx); sum += p8[v]; }
  sum += __shfl_xor(sum, 16, 32);
  const float inv = 1.0f / sum;

  // P^T as A fragment: elems 0..7 = P[g=half*8+e, h=l16] (matches acc layout),
  // elems 8..15 (K=16..31) zero-padded.
  uint4 alo = make_uint4(pk2(p8[0]*inv, p8[1]*inv), pk2(p8[2]*inv, p8[3]*inv),
                         pk2(p8[4]*inv, p8[5]*inv), pk2(p8[6]*inv, p8[7]*inv));
  v16bf aP = make_frag(alo, make_uint4(0,0,0,0));

#pragma unroll
  for (int j = 0; j < 4; ++j) {   // d tiles of 16
    uint4 blo = make_uint4(0,0,0,0), bhi = make_uint4(0,0,0,0);
    {                             // elems 0..7 = g=half*8+e, elems 8..15 (K=16..31) zero
      const unsigned short* p = &myvt[(j * 16 + l16) * 16 + half * 8];
      blo = *(const uint4*)p;
    }
    v8f o = {};
    o = __builtin_amdgcn_wmma_f32_16x16x32_bf16(false, aP, false, make_frag(blo, bhi),
                                                (short)0, o, false, false);
#pragma unroll
    for (int v = 0; v < 8; ++v)   // D: lane = d'=l16, VGPR v -> h = v + 8*half
      out[rowoff + (size_t)((v + 8 * half) * 64 + j * 16 + l16)] = bf_bits(o[v]);
  }
}

// ---------------------------------------------------------------------------
extern "C" void kernel_launch(void* const* d_in, const int* in_sizes, int n_in,
                              void* d_out, int out_size, void* d_ws, size_t ws_size,
                              hipStream_t stream) {
  (void)in_sizes; (void)n_in; (void)out_size; (void)ws_size;
  const float* content  = (const float*)d_in[0];
  const float* position = (const float*)d_in[1];
  const float* Wcq = (const float*)d_in[2];  const float* bcq = (const float*)d_in[3];
  const float* Wck = (const float*)d_in[4];  const float* bck = (const float*)d_in[5];
  const float* Wcv = (const float*)d_in[6];  const float* bcv = (const float*)d_in[7];
  const float* Wpq = (const float*)d_in[8];  const float* bpq = (const float*)d_in[9];
  const float* Wpk = (const float*)d_in[10]; const float* bpk = (const float*)d_in[11];
  const float* Wo  = (const float*)d_in[12]; const float* bo  = (const float*)d_in[13];

  const int M = 8 * 4096, N = 1024, K = 1024;
  const size_t SZ = (size_t)M * N;           // elements per bf16 intermediate

  unsigned short* cq = (unsigned short*)d_ws; // 6 x 64MB bf16 buffers in ws
  unsigned short* ck = cq + SZ;
  unsigned short* cv = ck + SZ;
  unsigned short* pq = cv + SZ;
  unsigned short* pk = pq + SZ;
  unsigned short* ao = pk + SZ;

  dim3 grid(N / BN, M / BM), block(256);
  gemm_bias_kernel<false, false><<<grid, block, 0, stream>>>(content,  Wcq, bcq, cq, M, N, K);
  gemm_bias_kernel<false, false><<<grid, block, 0, stream>>>(content,  Wck, bck, ck, M, N, K);
  gemm_bias_kernel<false, false><<<grid, block, 0, stream>>>(content,  Wcv, bcv, cv, M, N, K);
  gemm_bias_kernel<false, false><<<grid, block, 0, stream>>>(position, Wpq, bpq, pq, M, N, K);
  gemm_bias_kernel<false, false><<<grid, block, 0, stream>>>(position, Wpk, bpk, pk, M, N, K);

  attn_kernel<<<dim3(M / 8), block, 0, stream>>>(cq, ck, cv, pq, pk, ao);

  gemm_bias_kernel<true, true><<<grid, block, 0, stream>>>(ao, Wo, bo, d_out, M, N, K);
}
